// IndexNSW_17171279249868
// MI455X (gfx1250) — matrix operand order repaired and checked by reading, not compiled, not run on hardware
//
#include <hip/hip_runtime.h>

typedef __attribute__((ext_vector_type(16))) _Float16 v16h;
typedef __attribute__((ext_vector_type(8)))  float    v8f;

#define INF_F 1e30f

// ---------------------------------------------------------------------------
// Kernel 1: per-row squared norms of storage (streaming, HBM-bandwidth bound).
// One wave32 per row; 8 rows per 256-thread block.
// ---------------------------------------------------------------------------
__global__ __launch_bounds__(256) void nsw_norms_kernel(
    const float* __restrict__ storage, float* __restrict__ norms, int N) {
  int wave = threadIdx.x >> 5;
  int lane = threadIdx.x & 31;
  int row  = blockIdx.x * 8 + wave;
  if (row >= N) return;
  const float* s = storage + (size_t)row * 256;
  float acc = 0.f;
#pragma unroll
  for (int i = 0; i < 8; ++i) {
    float v = s[lane + i * 32];
    acc += v * v;
  }
#pragma unroll
  for (int off = 16; off > 0; off >>= 1) acc += __shfl_down(acc, off, 32);
  if (lane == 0) norms[row] = acc;
}

// ---------------------------------------------------------------------------
// Pack (dist, id, expanded) into a sortable u64 key. dist >= 0 so the raw f32
// bit pattern is monotonic as unsigned.
// ---------------------------------------------------------------------------
__device__ __forceinline__ unsigned long long packCand(float d, int id, int ex) {
  return ((unsigned long long)__float_as_uint(d) << 32) |
         ((unsigned int)id << 1) | (unsigned int)ex;
}
__device__ __forceinline__ int candId(unsigned long long e) {
  return (int)(((unsigned int)e) >> 1);
}
__device__ __forceinline__ float candDist(unsigned long long e) {
  return __uint_as_float((unsigned int)(e >> 32));
}

// Bitonic sort of 64 u64 keys in LDS, ascending. All 256 threads participate
// in control flow (uniform); threads 0..63 do the work.
__device__ __forceinline__ void bitonic64(unsigned long long* c, int tid) {
  for (int k = 2; k <= 64; k <<= 1) {
    for (int j = k >> 1; j > 0; j >>= 1) {
      __syncthreads();
      if (tid < 64) {
        int i = tid, l = i ^ j;
        if (l > i) {
          unsigned long long a = c[i], b = c[l];
          bool asc = ((i & k) == 0);
          if ((a > b) == asc) { c[i] = b; c[l] = a; }
        }
      }
    }
  }
  __syncthreads();
}

// ---------------------------------------------------------------------------
// Distance engine: 32 gathered rows vs query via v_wmma_f32_16x16x32_f16.
// wave&1   -> neighbor tile (rows 0-15 or 16-31)
// wave>>1  -> K-split (each wave covers 2 chunks of 32 dims = 64 dims)
// Per wave: 2 WMMAs; column 0 of the 16x16 result (lanes 0 and 16) holds the
// 16 dot products; partials summed into s_dot with ds_add_f32.
// Caller must have s_nbr populated, s_dot zeroed, and a barrier issued.
// ---------------------------------------------------------------------------
__device__ __forceinline__ void dist32_wmma(
    const float* __restrict__ storage, const int* s_nbr, float* s_dot,
    v16h b0, v16h b1, int tid) {
  int wave  = tid >> 5;
  int lane  = tid & 31;
  int tile  = wave & 1;
  int kbase = (wave >> 1) << 1;   // chunk indices kbase, kbase+1 (32 dims each)
  int hi    = lane >> 4;          // lane half selects K sub-range
  int id    = s_nbr[tile * 16 + (lane & 15)];
  const float* srow = storage + (size_t)id * 256;

  v8f acc = {};
  {
    // A-matrix 16x32 f16 layout: lane<16 halves = K{0..7,16..23}, lane>=16
    // halves = K{8..15,24..31} (shifted by hi*8).
    int d0 = kbase * 32 + hi * 8;
    int d1 = kbase * 32 + 16 + hi * 8;
    v16h a;
#pragma unroll
    for (int e = 0; e < 8; ++e) a[e]     = (_Float16)srow[d0 + e];
#pragma unroll
    for (int e = 0; e < 8; ++e) a[8 + e] = (_Float16)srow[d1 + e];
    acc = __builtin_amdgcn_wmma_f32_16x16x32_f16(false, a, false, b0,
                                                 (short)0, acc, false, false);
  }
  {
    int d0 = (kbase + 1) * 32 + hi * 8;
    int d1 = (kbase + 1) * 32 + 16 + hi * 8;
    v16h a;
#pragma unroll
    for (int e = 0; e < 8; ++e) a[e]     = (_Float16)srow[d0 + e];
#pragma unroll
    for (int e = 0; e < 8; ++e) a[8 + e] = (_Float16)srow[d1 + e];
    acc = __builtin_amdgcn_wmma_f32_16x16x32_f16(false, a, false, b1,
                                                 (short)0, acc, false, false);
  }
  // C/D layout: VGPR r -> row r (lanes 0-15) or r+8 (lanes 16-31); col = lane&15.
  if ((lane & 15) == 0) {
#pragma unroll
    for (int r = 0; r < 8; ++r)
      atomicAdd(&s_dot[tile * 16 + hi * 8 + r], acc[r]);
  }
}

// ---------------------------------------------------------------------------
// Kernel 2: one workgroup (8 waves) per query. Serial greedy search with the
// visited bitmap + pool in LDS; WMMA distances; bitonic top-32 merge.
// ---------------------------------------------------------------------------
__global__ __launch_bounds__(256) void nsw_search_kernel(
    const float* __restrict__ query, const float* __restrict__ storage,
    const int* __restrict__ graph, const float* __restrict__ norms,
    const int* __restrict__ initial, int* __restrict__ out_ids,
    float* __restrict__ out_d, int N, int R, int EF, int k, int visWords) {
  extern __shared__ unsigned int s_vis[];            // (N+31)/32 words
  __shared__ float              s_q[256];
  __shared__ _Float16           s_qh[256];
  __shared__ unsigned long long s_cand[64];          // [0..31]=pool, [32..63]=new
  __shared__ int                s_nbr[32];
  __shared__ float              s_dot[32];
  __shared__ float              s_nrm[32];
  __shared__ unsigned int       s_fresh;
  __shared__ int                s_u;
  __shared__ float              s_qn;

  const int tid = threadIdx.x;
  const int bid = blockIdx.x;

  // ---- init: load query, clear visited bitmap -----------------------------
  {
    float v = query[(size_t)bid * 256 + tid];
    s_q[tid]  = v;
    s_qh[tid] = (_Float16)v;
  }
  for (int i = tid; i < visWords; i += 256) s_vis[i] = 0u;
  __syncthreads();

  // ||q||^2 via wave 0
  if (tid < 32) {
    float a = 0.f;
#pragma unroll
    for (int i = 0; i < 8; ++i) { float v = s_q[tid + i * 32]; a += v * v; }
#pragma unroll
    for (int off = 16; off > 0; off >>= 1) a += __shfl_down(a, off, 32);
    if (tid == 0) s_qn = a;
  }

  // Per-wave B fragments (q replicated in all 16 columns; loop-invariant).
  // B 32x16 f16 layout: lanes 0-15 hold K {0..7,16..23}, lanes 16-31 hold K {8..15,24..31}.
  v16h b0, b1;
  {
    int wave = tid >> 5, lane = tid & 31, hi = lane >> 4;
    int kbase = (wave >> 1) << 1;
    int base0 = kbase * 32 + hi * 8;
    int base1 = (kbase + 1) * 32 + hi * 8;
#pragma unroll
    for (int e = 0; e < 16; ++e) {
      b0[e] = s_qh[base0 + (e < 8 ? e : e + 8)];
      b1[e] = s_qh[base1 + (e < 8 ? e : e + 8)];
    }
  }

  // ---- initial pool --------------------------------------------------------
  if (tid < 32) {
    int id = initial[(size_t)bid * EF + tid];
    s_nbr[tid] = id;
    s_nrm[tid] = norms[id];
    s_dot[tid] = 0.f;
    atomicOr(&s_vis[(unsigned)id >> 5], 1u << (id & 31));
  }
  __syncthreads();
  dist32_wmma(storage, s_nbr, s_dot, b0, b1, tid);
  __syncthreads();
  if (tid < 32) {
    float dist = fmaxf(s_nrm[tid] - 2.f * s_dot[tid] + s_qn, 0.f);
    s_cand[tid]      = packCand(dist, s_nbr[tid], 0);
    s_cand[32 + tid] = packCand(INF_F, 0, 0);
  }
  bitonic64(s_cand, tid);   // pool now sorted ascending in s_cand[0..31]

  // ---- serial greedy expansion --------------------------------------------
  for (int step = 0; step < EF; ++step) {
    if (tid == 0) {
      int j = 0;
      for (int i = 0; i < 32; ++i)
        if (!(s_cand[i] & 1ull)) { j = i; break; }
      int u = candId(s_cand[j]);
      s_u = u;
      // mark every pool entry holding this node as expanded
      for (int i = 0; i < 32; ++i)
        if (candId(s_cand[i]) == u) s_cand[i] |= 1ull;
      s_fresh = 0u;
    }
    __syncthreads();

    // gather neighbor ids + norms; read visited BEFORE any update (matches ref)
    if (tid < 32) {
      int nb = graph[(size_t)s_u * R + tid];
      s_nbr[tid] = nb;
      s_nrm[tid] = norms[nb];
      s_dot[tid] = 0.f;
      unsigned int w = s_vis[(unsigned)nb >> 5];
      if (!((w >> (nb & 31)) & 1u)) atomicOr(&s_fresh, 1u << tid);
    }
    __syncthreads();
    if (tid < 32) {
      int nb = s_nbr[tid];
      atomicOr(&s_vis[(unsigned)nb >> 5], 1u << (nb & 31));
    }
    __syncthreads();

    dist32_wmma(storage, s_nbr, s_dot, b0, b1, tid);
    __syncthreads();

    if (tid < 32) {
      float dist = fmaxf(s_nrm[tid] - 2.f * s_dot[tid] + s_qn, 0.f);
      bool fresh = (s_fresh >> tid) & 1u;
      s_cand[32 + tid] = packCand(fresh ? dist : INF_F, s_nbr[tid], 0);
    }
    bitonic64(s_cand, tid);  // keep best 32 as new pool
  }

  // ---- output: top-k (pool already sorted ascending) ----------------------
  if (tid < k) {
    unsigned long long e = s_cand[tid];
    out_ids[(size_t)bid * k + tid] = candId(e);
    out_d[(size_t)bid * k + tid]   = candDist(e);
  }
}

// ---------------------------------------------------------------------------
extern "C" void kernel_launch(void* const* d_in, const int* in_sizes, int n_in,
                              void* d_out, int out_size, void* d_ws,
                              size_t ws_size, hipStream_t stream) {
  const float* query   = (const float*)d_in[0];
  const float* storage = (const float*)d_in[1];
  const int*   graph   = (const int*)d_in[2];
  const int*   initial = (const int*)d_in[3];
  // k (d_in[4]) and ef_search (d_in[5]) are scalars; derive on host instead
  // (graph-capture safe, no device reads):
  const int D  = 256;
  const int B  = in_sizes[0] / D;
  const int N  = in_sizes[1] / D;
  const int R  = in_sizes[2] / N;
  const int EF = in_sizes[3] / B;          // == ef_search in the reference setup
  const int k  = out_size / (2 * B);       // ids + dists concatenated

  float* norms   = (float*)d_ws;           // N floats of scratch
  int*   out_ids = (int*)d_out;            // first B*k: int32 ids
  float* out_d   = (float*)d_out + (size_t)B * k;  // next B*k: f32 dists

  nsw_norms_kernel<<<(N + 7) / 8, 256, 0, stream>>>(storage, norms, N);

  const int    visWords = (N + 31) / 32;
  const size_t shmem    = (size_t)visWords * sizeof(unsigned int);
  nsw_search_kernel<<<B, 256, shmem, stream>>>(query, storage, graph, norms,
                                               initial, out_ids, out_d, N, R,
                                               EF, k, visWords);
}
